// Model_18253611008007
// MI455X (gfx1250) — matrix operand order repaired and checked by reading, not compiled, not run on hardware
//
#include <hip/hip_runtime.h>
#include <hip/hip_bf16.h>
#include <math.h>

// ---------------------------------------------------------------------------
// Types for CDNA5 WMMA (wave32, v_wmma_f32_16x16x32_f16)
// ---------------------------------------------------------------------------
typedef __attribute__((ext_vector_type(16))) _Float16 v16h;
typedef __attribute__((ext_vector_type(8)))  _Float16 v8h;
typedef __attribute__((ext_vector_type(8)))  float    v8f;

#define HH 32
#define BB 512
#define HBR (HH * BB)
#define NLL 10
#define NRR 11

// ---------------------------------------------------------------------------
// Small device helpers
// ---------------------------------------------------------------------------
__device__ __forceinline__ float wsum32(float v) {
#pragma unroll
  for (int off = 16; off; off >>= 1) v += __shfl_xor(v, off, 32);
  return v;
}
__device__ __forceinline__ float lrelu_f(float x) { return x > 0.f ? x : 0.01f * x; }
__device__ __forceinline__ float elu_f(float x)   { return x > 0.f ? x : expf(x) - 1.f; }
__device__ __forceinline__ float sigm_f(float x)  { return 1.f / (1.f + expf(-x)); }

// ---------------------------------------------------------------------------
// Generic WMMA GEMM, register-blocked 16(M) x 64(N) per wave:
//   C[M,N] = A[M,K] (f16, row-major, lda) * W[N,K]^T (f16, row-major over K,
//   ldb) + bias0 + bias1.
// One wave per block; 4 accumulators share one A fragment (4x A reuse).
// Per k-step: issue ALL fragment loads first, then 4 back-to-back WMMAs so
// the scheduler needs a single loadcnt wait per step.
// M mult of 16; N mult of 64; K mult of 32.  Bias applied only for n<Nvalid.
// ---------------------------------------------------------------------------
__global__ void gemm_f16_wmma(const _Float16* __restrict__ A, int lda,
                              const _Float16* __restrict__ B, int ldb,
                              float* __restrict__ C, int ldc,
                              const float* __restrict__ bias0,
                              const float* __restrict__ bias1,
                              int K, int tilesN64, int Nvalid) {
  const int tile = blockIdx.x;
  const int tn = tile % tilesN64;  // 64-wide N tile
  const int tm = tile / tilesN64;
  const int lane = threadIdx.x & 31;

  const int aRow     = tm * 16 + (lane & 15);
  const int aKoff    = (lane >> 4) << 3;   // 0 or 8  (ISA A-matrix 16-bit layout)
  const int bColBase = tn * 64 + (lane & 15);
  const int bKoff    = (lane >> 4) << 3;   // 0 or 8  (ISA B-matrix 16-bit layout)

  const _Float16* ap  = A + (long)aRow * lda + aKoff;
  const _Float16* bp0 = B + (long)bColBase * ldb + bKoff;

  v8f acc[4] = {};
  for (int kk = 0; kk < K; kk += 32) {
    if (kk + 32 < K) {
      __builtin_prefetch(ap + kk + 32, 0, 3);  // near-cache prefetch of next slab
#pragma unroll
      for (int s = 0; s < 4; s++)
        __builtin_prefetch(bp0 + (long)s * 16 * ldb + kk + 32, 0, 3);
    }
    // ---- issue all loads for this k-step ----
    v8h a_lo = *(const v8h*)(ap + kk);       // K = kk+koff+0..7
    v8h a_hi = *(const v8h*)(ap + kk + 16);  // K = kk+koff+16..23
    v8h b_lo[4], b_hi[4];
#pragma unroll
    for (int s = 0; s < 4; s++) {
      const _Float16* bp = bp0 + (long)s * 16 * ldb;
      b_lo[s] = *(const v8h*)(bp + kk);
      b_hi[s] = *(const v8h*)(bp + kk + 16);
    }
    // ---- then compute: 4 back-to-back WMMAs sharing one A fragment ----
    v16h a = __builtin_shufflevector(a_lo, a_hi,
                                     0,1,2,3,4,5,6,7,8,9,10,11,12,13,14,15);
#pragma unroll
    for (int s = 0; s < 4; s++) {
      v16h b = __builtin_shufflevector(b_lo[s], b_hi[s],
                                       0,1,2,3,4,5,6,7,8,9,10,11,12,13,14,15);
      acc[s] = __builtin_amdgcn_wmma_f32_16x16x32_f16(false, a, false, b,
                                                      (short)0, acc[s],
                                                      false, false);
    }
  }

  const int rbase = tm * 16 + ((lane >> 4) << 3);
#pragma unroll
  for (int s = 0; s < 4; s++) {
    const int bCol = bColBase + s * 16;
    float bsum = 0.f;
    if (bCol < Nvalid) {
      if (bias0) bsum += bias0[bCol];
      if (bias1) bsum += bias1[bCol];
    }
#pragma unroll
    for (int v = 0; v < 8; v++)
      C[(long)(rbase + v) * ldc + bCol] = acc[s][v] + bsum;
  }
}

// ---------------------------------------------------------------------------
// Fused embedding: y = relu(LN(x @ W^T + b)); 64 outputs; one block per row.
// Writes f16 into Y (possibly a slice of the 576-wide cat buffer).
// ---------------------------------------------------------------------------
__global__ void embed_k(const float* __restrict__ X, int K,
                        const float* __restrict__ W, const float* __restrict__ b,
                        const float* __restrict__ g, const float* __restrict__ e,
                        _Float16* __restrict__ Y, int ldy) {
  const int row = blockIdx.x;
  const int t = threadIdx.x;  // 64 threads
  __shared__ float red[64];
  const float* x = X + (long)row * K;
  const float* w = W + (long)t * K;
  float acc = b[t];
  for (int k = 0; k < K; k++) acc += x[k] * w[k];
  red[t] = acc; __syncthreads();
  for (int s = 32; s > 0; s >>= 1) { if (t < s) red[t] += red[t + s]; __syncthreads(); }
  float mean = red[0] * (1.f / 64.f); __syncthreads();
  float d = acc - mean;
  red[t] = d * d; __syncthreads();
  for (int s = 32; s > 0; s >>= 1) { if (t < s) red[t] += red[t + s]; __syncthreads(); }
  float var = red[0] * (1.f / 64.f);
  float y = fmaxf(g[t] * d * rsqrtf(var + 1e-5f) + e[t], 0.f);
  Y[(long)row * ldy + t] = (_Float16)y;
}

// ---------------------------------------------------------------------------
// att_self: query vs [self]+team.  One wave per HB row; dims lane & lane+32.
// qw [HB,64], kw [HB*NK,64] (W-transformed, fp32), team_h raw f16 embeddings.
// agg_h  <- elu(softmax_{1+NK} combo of qw,kw)              (f16, feeds GEMM)
// catsl  <- sum softmax(s[1:]) * raw team embedding         (f16 cat slice)
// ---------------------------------------------------------------------------
template <int NK>
__global__ void att_self_k(const float* __restrict__ qw, const float* __restrict__ kw,
                           const _Float16* __restrict__ team_h,
                           const float* __restrict__ av, const float* __restrict__ ba,
                           _Float16* __restrict__ agg_h,
                           _Float16* __restrict__ catsl, int ldcat, int total) {
  const int gw = blockIdx.x * (blockDim.x >> 5) + (threadIdx.x >> 5);
  const int lane = threadIdx.x & 31;
  if (gw >= total) return;
  const float bas = ba[0];
  const float q0 = qw[(long)gw * 64 + lane];
  const float q1 = qw[(long)gw * 64 + lane + 32];
  const float sq_aq = wsum32(q0 * av[lane] + q1 * av[lane + 32]);
  const float sq_ak = wsum32(q0 * av[64 + lane] + q1 * av[96 + lane]);
  float s[NK + 1];
  s[0] = lrelu_f(sq_aq + sq_ak + bas);
  float k0[NK], k1[NK];
#pragma unroll
  for (int j = 0; j < NK; j++) {
    long off = ((long)gw * NK + j) * 64;
    k0[j] = kw[off + lane]; k1[j] = kw[off + lane + 32];
    s[j + 1] = lrelu_f(sq_aq + wsum32(k0[j] * av[64 + lane] + k1[j] * av[96 + lane]) + bas);
  }
  float mx = s[0];
#pragma unroll
  for (int j = 1; j <= NK; j++) mx = fmaxf(mx, s[j]);
  float w[NK + 1], den = 0.f;
#pragma unroll
  for (int j = 0; j <= NK; j++) { w[j] = expf(s[j] - mx); den += w[j]; }
  const float inv = 1.f / den;
  float g0 = w[0] * inv * q0, g1 = w[0] * inv * q1;
#pragma unroll
  for (int j = 0; j < NK; j++) { float aj = w[j + 1] * inv; g0 += aj * k0[j]; g1 += aj * k1[j]; }
  agg_h[(long)gw * 64 + lane]      = (_Float16)elu_f(g0);
  agg_h[(long)gw * 64 + lane + 32] = (_Float16)elu_f(g1);

  float mx2 = s[1];
#pragma unroll
  for (int j = 2; j <= NK; j++) mx2 = fmaxf(mx2, s[j]);
  float w2[NK], den2 = 0.f;
#pragma unroll
  for (int j = 0; j < NK; j++) { w2[j] = expf(s[j + 1] - mx2); den2 += w2[j]; }
  float r0 = 0.f, r1 = 0.f;
#pragma unroll
  for (int j = 0; j < NK; j++) {
    long off = ((long)gw * NK + j) * 64;
    float aj = w2[j] / den2;
    r0 += aj * (float)team_h[off + lane];
    r1 += aj * (float)team_h[off + lane + 32];
  }
  catsl[(long)gw * ldcat + lane]      = (_Float16)r0;
  catsl[(long)gw * ldcat + lane + 32] = (_Float16)r1;
}

// ---------------------------------------------------------------------------
// att_team: every query player vs [self]+opposing team.  One wave per (row,q).
// ---------------------------------------------------------------------------
template <int NK>
__global__ void att_team_k(const float* __restrict__ qwAll, const float* __restrict__ kwAll,
                           int NQ, const float* __restrict__ av, const float* __restrict__ ba,
                           _Float16* __restrict__ out_h, int total) {
  const int gw = blockIdx.x * (blockDim.x >> 5) + (threadIdx.x >> 5);
  const int lane = threadIdx.x & 31;
  if (gw >= total) return;
  const int hb = gw / NQ;
  const float bas = ba[0];
  const float q0 = qwAll[(long)gw * 64 + lane];
  const float q1 = qwAll[(long)gw * 64 + lane + 32];
  const float s0 = wsum32(q0 * av[lane] + q1 * av[lane + 32]);
  float s[NK + 1];
  s[0] = lrelu_f(s0 + wsum32(q0 * av[64 + lane] + q1 * av[96 + lane]) + bas);
  float k0[NK], k1[NK];
#pragma unroll
  for (int k = 0; k < NK; k++) {
    long off = ((long)hb * NK + k) * 64;
    k0[k] = kwAll[off + lane]; k1[k] = kwAll[off + lane + 32];
    s[k + 1] = lrelu_f(s0 + wsum32(k0[k] * av[64 + lane] + k1[k] * av[96 + lane]) + bas);
  }
  float mx = s[0];
#pragma unroll
  for (int j = 1; j <= NK; j++) mx = fmaxf(mx, s[j]);
  float w[NK + 1], den = 0.f;
#pragma unroll
  for (int j = 0; j <= NK; j++) { w[j] = expf(s[j] - mx); den += w[j]; }
  const float inv = 1.f / den;
  float g0 = w[0] * inv * q0, g1 = w[0] * inv * q1;
#pragma unroll
  for (int k = 0; k < NK; k++) { float ak = w[k + 1] * inv; g0 += ak * k0[k]; g1 += ak * k1[k]; }
  out_h[(long)gw * 64 + lane]      = (_Float16)elu_f(g0);
  out_h[(long)gw * 64 + lane + 32] = (_Float16)elu_f(g1);
}

// ---------------------------------------------------------------------------
// att2: scores only (no self slot), then weighted sum of RAW team embeddings
// into a cat slice.  One wave per HB row.
// ---------------------------------------------------------------------------
template <int NK>
__global__ void att2_k(const float* __restrict__ qw2, const float* __restrict__ kw2,
                       const float* __restrict__ av, const float* __restrict__ ba,
                       const _Float16* __restrict__ team_h,
                       _Float16* __restrict__ catsl, int ldcat, int total) {
  const int gw = blockIdx.x * (blockDim.x >> 5) + (threadIdx.x >> 5);
  const int lane = threadIdx.x & 31;
  if (gw >= total) return;
  const float bas = ba[0];
  const float q0 = qw2[(long)gw * 64 + lane];
  const float q1 = qw2[(long)gw * 64 + lane + 32];
  const float sq = wsum32(q0 * av[lane] + q1 * av[lane + 32]);
  float s[NK];
#pragma unroll
  for (int k = 0; k < NK; k++) {
    long off = ((long)gw * NK + k) * 64;
    float c0 = kw2[off + lane], c1 = kw2[off + lane + 32];
    s[k] = lrelu_f(sq + wsum32(c0 * av[64 + lane] + c1 * av[96 + lane]) + bas);
  }
  float mx = s[0];
#pragma unroll
  for (int j = 1; j < NK; j++) mx = fmaxf(mx, s[j]);
  float w[NK], den = 0.f;
#pragma unroll
  for (int j = 0; j < NK; j++) { w[j] = expf(s[j] - mx); den += w[j]; }
  float r0 = 0.f, r1 = 0.f;
#pragma unroll
  for (int k = 0; k < NK; k++) {
    long off = ((long)gw * NK + k) * 64;
    float ak = w[k] / den;
    r0 += ak * (float)team_h[off + lane];
    r1 += ak * (float)team_h[off + lane + 32];
  }
  catsl[(long)gw * ldcat + lane]      = (_Float16)r0;
  catsl[(long)gw * ldcat + lane + 32] = (_Float16)r1;
}

// ---------------------------------------------------------------------------
// LN(256)+ReLU -> f16 (for x feeding the LSTM GEMMs).  Block = 256 = row.
// ---------------------------------------------------------------------------
__global__ void ln_relu_256(const float* __restrict__ X, const float* __restrict__ g,
                            const float* __restrict__ e, _Float16* __restrict__ Y) {
  const int row = blockIdx.x, t = threadIdx.x;
  __shared__ float red[256];
  float v = X[(long)row * 256 + t];
  red[t] = v; __syncthreads();
  for (int s = 128; s > 0; s >>= 1) { if (t < s) red[t] += red[t + s]; __syncthreads(); }
  float mean = red[0] * (1.f / 256.f); __syncthreads();
  float d = v - mean;
  red[t] = d * d; __syncthreads();
  for (int s = 128; s > 0; s >>= 1) { if (t < s) red[t] += red[t + s]; __syncthreads(); }
  float var = red[0] * (1.f / 256.f);
  Y[(long)row * 256 + t] = (_Float16)fmaxf(g[t] * d * rsqrtf(var + 1e-5f) + e[t], 0.f);
}

// ---------------------------------------------------------------------------
// LSTM pointwise gates for one time step.
// ---------------------------------------------------------------------------
__global__ void lstm_point(const float* __restrict__ gih, const float* __restrict__ ghh,
                           float* __restrict__ c, _Float16* __restrict__ h_h,
                           _Float16* __restrict__ out_h) {
  const int i = blockIdx.x * blockDim.x + threadIdx.x;
  if (i >= BB * 256) return;
  const int b = i >> 8, j = i & 255;
  const float* gi = gih + (long)b * 1024;
  const float* gh = ghh + (long)b * 1024;
  float ig = gi[j]       + gh[j];
  float fg = gi[256 + j] + gh[256 + j];
  float gg = gi[512 + j] + gh[512 + j];
  float og = gi[768 + j] + gh[768 + j];
  float cc = sigm_f(fg) * c[i] + sigm_f(ig) * tanhf(gg);
  float hh = sigm_f(og) * tanhf(cc);
  c[i] = cc;
  h_h[i]   = (_Float16)hh;
  out_h[i] = (_Float16)hh;
}

// ---------------------------------------------------------------------------
// Head finalize: LN(164)+ReLU for a/m/v (stored with ld=192), tiny output
// GEMVs, softmaxes.  One 256-thread block per HB row; writes out[row, 0:21].
// ---------------------------------------------------------------------------
__global__ void heads_final(const float* __restrict__ a1, const float* __restrict__ m1,
                            const float* __restrict__ v1,
                            const float* g_a1, const float* e_a1,
                            const float* Wa2, const float* ba2,
                            const float* g_m1, const float* e_m1,
                            const float* Wm2, const float* bm2,
                            const float* g_v1, const float* e_v1,
                            const float* Wv2, const float* __restrict__ avail,
                            float* __restrict__ out) {
  const int row = blockIdx.x, t = threadIdx.x;
  __shared__ float red[256];
  __shared__ float buf[176];
  __shared__ float lg[16];
  float* o = out + (long)row * 21;

  auto lnrelu = [&](const float* src, const float* g, const float* e) {
    float v = (t < 164) ? src[(long)row * 192 + t] : 0.f;
    red[t] = v; __syncthreads();
    for (int s = 128; s > 0; s >>= 1) { if (t < s) red[t] += red[t + s]; __syncthreads(); }
    float mean = red[0] * (1.f / 164.f); __syncthreads();
    float d = (t < 164) ? v - mean : 0.f;
    red[t] = d * d; __syncthreads();
    for (int s = 128; s > 0; s >>= 1) { if (t < s) red[t] += red[t + s]; __syncthreads(); }
    float var = red[0] * (1.f / 164.f); __syncthreads();
    if (t < 176) buf[t] = (t < 164) ? fmaxf(g[t] * d * rsqrtf(var + 1e-5f) + e[t], 0.f) : 0.f;
    __syncthreads();
  };

  // action head (12) with availability mask
  lnrelu(a1, g_a1, e_a1);
  if (t < 12) {
    float acc = ba2[t];
    for (int k = 0; k < 164; k++) acc += buf[k] * Wa2[t * 164 + k];
    acc += (avail[(long)row * 12 + t] - 1.f) * 1e7f;
    lg[t] = acc;
  }
  __syncthreads();
  if (t == 0) {
    float mx = lg[0]; for (int j = 1; j < 12; j++) mx = fmaxf(mx, lg[j]);
    float s = 0.f, ev[12];
    for (int j = 0; j < 12; j++) { ev[j] = expf(lg[j] - mx); s += ev[j]; }
    for (int j = 0; j < 12; j++) o[j] = ev[j] / s;
  }
  __syncthreads();
  // move head (8)
  lnrelu(m1, g_m1, e_m1);
  if (t < 8) {
    float acc = bm2[t];
    for (int k = 0; k < 164; k++) acc += buf[k] * Wm2[t * 164 + k];
    lg[t] = acc;
  }
  __syncthreads();
  if (t == 0) {
    float mx = lg[0]; for (int j = 1; j < 8; j++) mx = fmaxf(mx, lg[j]);
    float s = 0.f, ev[8];
    for (int j = 0; j < 8; j++) { ev[j] = expf(lg[j] - mx); s += ev[j]; }
    for (int j = 0; j < 8; j++) o[12 + j] = ev[j] / s;
  }
  __syncthreads();
  // value head (1, no bias on Wv2)
  lnrelu(v1, g_v1, e_v1);
  if (t == 0) {
    float acc = 0.f;
    for (int k = 0; k < 164; k++) acc += buf[k] * Wv2[k];
    o[20] = acc;
  }
}

// ---------------------------------------------------------------------------
// Conversion utilities
// ---------------------------------------------------------------------------
__global__ void f32_to_f16_k(const float* __restrict__ s, _Float16* __restrict__ d, long n) {
  long i = (long)blockIdx.x * blockDim.x + threadIdx.x;
  if (i < n) d[i] = (_Float16)s[i];
}
__global__ void f32_to_f16_pad_k(const float* __restrict__ s, _Float16* __restrict__ d,
                                 int rows, int cols, int orows) {
  long i = (long)blockIdx.x * blockDim.x + threadIdx.x;
  long tot = (long)orows * cols;
  if (i >= tot) return;
  int r = (int)(i / cols), c = (int)(i % cols);
  d[i] = (r < rows) ? (_Float16)s[(long)r * cols + c] : (_Float16)0.f;
}

// ---------------------------------------------------------------------------
// Host side
// ---------------------------------------------------------------------------
enum {
  IN_MS = 0, IN_PS, IN_BS, IN_PE, IN_BE, IN_LE, IN_RE, IN_AVAIL, IN_H0, IN_C0,
  PW_EMB = 10,                           // 7 x {W,b,g,e} -> idx 10..37
  P_W1AW = 38, P_B1AW, P_W2AW, P_B2AW, P_W1DW, P_B1DW, P_W2DW, P_B2DW,
  P_A1AA = 46, P_BA1AA, P_A2AA, P_BA2AA, P_A1DA, P_BA1DA, P_A2DA, P_BA2DA,
  P_WCAT = 54, P_BCAT, P_GCAT, P_ECAT,
  P_WIH = 58, P_WHH, P_BIH, P_BHH,
  P_WA1 = 62, P_BA1, P_GA1, P_EA1, P_WA2, P_BA2,
  P_WM1 = 68, P_BM1, P_GM1, P_EM1, P_WM2, P_BM2,
  P_WV1 = 74, P_BV1, P_GV1, P_EV1, P_WV2
};

static inline void launch_gemm(hipStream_t s, const _Float16* A, int lda,
                               const _Float16* B, int ldb, float* C, int ldc,
                               const float* b0, const float* b1,
                               int M, int N, int K, int Nvalid) {
  int tiles = (M / 16) * (N / 64);
  gemm_f16_wmma<<<tiles, 32, 0, s>>>(A, lda, B, ldb, C, ldc, b0, b1, K, N / 64, Nvalid);
}

extern "C" void kernel_launch(void* const* d_in, const int* in_sizes, int n_in,
                              void* d_out, int out_size, void* d_ws, size_t ws_size,
                              hipStream_t stream) {
  (void)in_sizes; (void)n_in; (void)out_size; (void)ws_size;
  const float* prm[96];
  for (int i = 0; i < n_in && i < 96; i++) prm[i] = (const float*)d_in[i];
  float* out = (float*)d_out;

  // ---- workspace bump allocator (256B aligned) ----
  char* wsp = (char*)d_ws;
  auto alloc = [&](size_t bytes) -> char* {
    char* r = wsp; wsp += (bytes + 255) & ~(size_t)255; return r;
  };
  // f16 activations
  _Float16* cat_h = (_Float16*)alloc((size_t)HBR * 576 * 2);
  _Float16* le_h  = (_Float16*)alloc((size_t)HBR * NLL * 64 * 2);
  _Float16* re_h  = (_Float16*)alloc((size_t)HBR * NRR * 64 * 2);
  _Float16* pre_h = (_Float16*)alloc((size_t)HBR * 64 * 2);
  _Float16* ble_h = (_Float16*)alloc((size_t)HBR * 64 * 2);
  _Float16* lre_h = (_Float16*)alloc((size_t)HBR * NLL * 64 * 2);
  _Float16* rle_h = (_Float16*)alloc((size_t)HBR * NRR * 64 * 2);
  _Float16* x_h   = (_Float16*)alloc((size_t)HBR * 256 * 2);
  _Float16* h_h   = (_Float16*)alloc((size_t)BB * 256 * 2);
  _Float16* out_h = (_Float16*)alloc((size_t)HBR * 256 * 2);
  // f16 weights
  _Float16* w1aw_h = (_Float16*)alloc((size_t)64 * 64 * 2);
  _Float16* w2aw_h = (_Float16*)alloc((size_t)64 * 64 * 2);
  _Float16* w1dw_h = (_Float16*)alloc((size_t)64 * 64 * 2);
  _Float16* w2dw_h = (_Float16*)alloc((size_t)64 * 64 * 2);
  _Float16* wcat_h = (_Float16*)alloc((size_t)256 * 576 * 2);
  _Float16* wih_h  = (_Float16*)alloc((size_t)1024 * 256 * 2);
  _Float16* whh_h  = (_Float16*)alloc((size_t)1024 * 256 * 2);
  _Float16* wa1_h  = (_Float16*)alloc((size_t)192 * 256 * 2);
  _Float16* wm1_h  = (_Float16*)alloc((size_t)192 * 256 * 2);
  _Float16* wv1_h  = (_Float16*)alloc((size_t)192 * 256 * 2);
  // fp32 intermediates
  float* peA  = (float*)alloc((size_t)HBR * 64 * 4);
  float* beD  = (float*)alloc((size_t)HBR * 64 * 4);
  float* leA  = (float*)alloc((size_t)HBR * NLL * 64 * 4);
  float* reA  = (float*)alloc((size_t)HBR * NRR * 64 * 4);
  float* leD  = (float*)alloc((size_t)HBR * NLL * 64 * 4);
  float* reD  = (float*)alloc((size_t)HBR * NRR * 64 * 4);
  float* qA2  = (float*)alloc((size_t)HBR * 64 * 4);
  float* kA2  = (float*)alloc((size_t)HBR * NLL * 64 * 4);
  float* qD2  = (float*)alloc((size_t)HBR * 64 * 4);
  float* kD2  = (float*)alloc((size_t)HBR * NRR * 64 * 4);
  float* xlin = (float*)alloc((size_t)HBR * 256 * 4);
  float* gih  = (float*)alloc((size_t)HBR * 1024 * 4);
  float* ghh  = (float*)alloc((size_t)BB * 1024 * 4);
  float* cst  = (float*)alloc((size_t)BB * 256 * 4);
  float* a1   = (float*)alloc((size_t)HBR * 192 * 4);
  float* m1   = (float*)alloc((size_t)HBR * 192 * 4);
  float* v1   = (float*)alloc((size_t)HBR * 192 * 4);

  // ---- weight conversions (fp32 -> f16; heads padded 164 -> 192 rows) ----
  auto cvt = [&](const float* s, _Float16* d, long n) {
    f32_to_f16_k<<<(int)((n + 255) / 256), 256, 0, stream>>>(s, d, n);
  };
  cvt(prm[P_W1AW], w1aw_h, 64 * 64);
  cvt(prm[P_W2AW], w2aw_h, 64 * 64);
  cvt(prm[P_W1DW], w1dw_h, 64 * 64);
  cvt(prm[P_W2DW], w2dw_h, 64 * 64);
  cvt(prm[P_WCAT], wcat_h, 256 * 576);
  cvt(prm[P_WIH],  wih_h, 1024 * 256);
  cvt(prm[P_WHH],  whh_h, 1024 * 256);
  f32_to_f16_pad_k<<<192, 256, 0, stream>>>(prm[P_WA1], wa1_h, 164, 256, 192);
  f32_to_f16_pad_k<<<192, 256, 0, stream>>>(prm[P_WM1], wm1_h, 164, 256, 192);
  f32_to_f16_pad_k<<<192, 256, 0, stream>>>(prm[P_WV1], wv1_h, 164, 256, 192);

  // ---- embeddings (fused Linear+LN+ReLU), writing f16 into cat / team bufs ----
  struct EmbCfg { int in, pb, K; _Float16* dst; int ld; int M; };
  EmbCfg ec[7] = {
    {IN_MS, PW_EMB + 0,  19, cat_h + 0,   576, HBR},
    {IN_PS, PW_EMB + 4,  29, cat_h + 64,  576, HBR},
    {IN_BS, PW_EMB + 8,  18, cat_h + 128, 576, HBR},
    {IN_PE, PW_EMB + 12, 29, cat_h + 192, 576, HBR},
    {IN_BE, PW_EMB + 16, 18, cat_h + 256, 576, HBR},
    {IN_LE, PW_EMB + 20, 23, le_h,        64,  HBR * NLL},
    {IN_RE, PW_EMB + 24, 23, re_h,        64,  HBR * NRR},
  };
  for (int i = 0; i < 7; i++)
    embed_k<<<ec[i].M, 64, 0, stream>>>(prm[ec[i].in], ec[i].K,
                                        prm[ec[i].pb], prm[ec[i].pb + 1],
                                        prm[ec[i].pb + 2], prm[ec[i].pb + 3],
                                        ec[i].dst, ec[i].ld);

  // ---- attention linears (WMMA, K=64,N=64) ----
  const _Float16* pe_h = cat_h + 192;  // lda 576 slices
  const _Float16* be_h = cat_h + 256;
  launch_gemm(stream, pe_h, 576, w1aw_h, 64, peA, 64, prm[P_B1AW], nullptr, HBR,       64, 64, 64);
  launch_gemm(stream, le_h, 64,  w1aw_h, 64, leA, 64, prm[P_B1AW], nullptr, HBR * NLL, 64, 64, 64);
  launch_gemm(stream, re_h, 64,  w1aw_h, 64, reA, 64, prm[P_B1AW], nullptr, HBR * NRR, 64, 64, 64);
  launch_gemm(stream, be_h, 576, w1dw_h, 64, beD, 64, prm[P_B1DW], nullptr, HBR,       64, 64, 64);
  launch_gemm(stream, le_h, 64,  w1dw_h, 64, leD, 64, prm[P_B1DW], nullptr, HBR * NLL, 64, 64, 64);
  launch_gemm(stream, re_h, 64,  w1dw_h, 64, reD, 64, prm[P_B1DW], nullptr, HBR * NRR, 64, 64, 64);

  // ---- stage-1 attention ----
  const int WPB = 4;  // waves per 128-thread block
  att_self_k<NRR><<<(HBR + WPB - 1) / WPB, WPB * 32, 0, stream>>>(
      peA, reA, re_h, prm[P_A1AA], prm[P_BA1AA], pre_h, cat_h + 384, 576, HBR);
  att_self_k<NLL><<<(HBR + WPB - 1) / WPB, WPB * 32, 0, stream>>>(
      beD, leD, le_h, prm[P_A1DA], prm[P_BA1DA], ble_h, cat_h + 448, 576, HBR);
  att_team_k<NLL><<<(HBR * NRR + WPB - 1) / WPB, WPB * 32, 0, stream>>>(
      reD, leD, NRR, prm[P_A1DA], prm[P_BA1DA], rle_h, HBR * NRR);   // right_left
  att_team_k<NRR><<<(HBR * NLL + WPB - 1) / WPB, WPB * 32, 0, stream>>>(
      leA, reA, NLL, prm[P_A1AA], prm[P_BA1AA], lre_h, HBR * NLL);   // left_right

  // ---- stage-2 linears (WMMA) + att2 -> remaining cat slices ----
  launch_gemm(stream, pre_h, 64, w2aw_h, 64, qA2, 64, prm[P_B2AW], nullptr, HBR,       64, 64, 64);
  launch_gemm(stream, lre_h, 64, w2aw_h, 64, kA2, 64, prm[P_B2AW], nullptr, HBR * NLL, 64, 64, 64);
  launch_gemm(stream, ble_h, 64, w2dw_h, 64, qD2, 64, prm[P_B2DW], nullptr, HBR,       64, 64, 64);
  launch_gemm(stream, rle_h, 64, w2dw_h, 64, kD2, 64, prm[P_B2DW], nullptr, HBR * NRR, 64, 64, 64);
  att2_k<NLL><<<(HBR + WPB - 1) / WPB, WPB * 32, 0, stream>>>(
      qA2, kA2, prm[P_A2AA], prm[P_BA2AA], le_h, cat_h + 320, 576, HBR);  // left_player
  att2_k<NRR><<<(HBR + WPB - 1) / WPB, WPB * 32, 0, stream>>>(
      qD2, kD2, prm[P_A2DA], prm[P_BA2DA], re_h, cat_h + 512, 576, HBR);  // right_ball

  // ---- cat GEMM (16384x256x576) + LN + ReLU ----
  launch_gemm(stream, cat_h, 576, wcat_h, 576, xlin, 256, prm[P_BCAT], nullptr, HBR, 256, 576, 256);
  ln_relu_256<<<HBR, 256, 0, stream>>>(xlin, prm[P_GCAT], prm[P_ECAT], x_h);

  // ---- LSTM: hoisted input GEMM, then 32 sequential (hh GEMM + gates) ----
  launch_gemm(stream, x_h, 256, wih_h, 256, gih, 1024, prm[P_BIH], prm[P_BHH], HBR, 1024, 256, 1024);
  f32_to_f16_k<<<(BB * 256 + 255) / 256, 256, 0, stream>>>(prm[IN_H0], h_h, (long)BB * 256);
  hipMemcpyAsync(cst, prm[IN_C0], (size_t)BB * 256 * 4, hipMemcpyDeviceToDevice, stream);
  for (int t = 0; t < HH; t++) {
    launch_gemm(stream, h_h, 256, whh_h, 256, ghh, 1024, nullptr, nullptr, BB, 1024, 256, 1024);
    lstm_point<<<(BB * 256 + 255) / 256, 256, 0, stream>>>(
        gih + (size_t)t * BB * 1024, ghh, cst, h_h, out_h + (size_t)t * BB * 256);
  }

  // ---- heads: WMMA first layers (N padded to 192), then finalize ----
  launch_gemm(stream, out_h, 256, wa1_h, 256, a1, 192, prm[P_BA1], nullptr, HBR, 192, 256, 164);
  launch_gemm(stream, out_h, 256, wm1_h, 256, m1, 192, prm[P_BM1], nullptr, HBR, 192, 256, 164);
  launch_gemm(stream, out_h, 256, wv1_h, 256, v1, 192, prm[P_BV1], nullptr, HBR, 192, 256, 164);
  heads_final<<<HBR, 256, 0, stream>>>(a1, m1, v1,
                                       prm[P_GA1], prm[P_EA1], prm[P_WA2], prm[P_BA2],
                                       prm[P_GM1], prm[P_EM1], prm[P_WM2], prm[P_BM2],
                                       prm[P_GV1], prm[P_EV1], prm[P_WV2],
                                       prm[IN_AVAIL], out);
}
